// Qwen3_5Block_41669772705848
// MI455X (gfx1250) — compile-verified, matrix operand order checked by reading, not run on hardware
//
#include <hip/hip_runtime.h>

// Gated delta rule (chunked), bf16 WMMA on gfx1250 wave32.
// Phase 1: per-chunk precompute (A, Tinv, u, kcd)  -- grid 2048 blocks.
// Phase 2: per-(b,h) sequential chunk scan with async-to-LDS double buffering
//          of the (u, kcd) stage + prefetch of next chunk q/k -- grid 32 blocks.
// Workspace: u (bf16) + kcd (bf16) = 64 MB in d_ws.

typedef __attribute__((ext_vector_type(16))) __bf16 bf16x16;
typedef __attribute__((ext_vector_type(8)))  float  f32x8;
typedef int v4i __attribute__((vector_size(16)));            // async-LDS builtin operand type
typedef __attribute__((address_space(1))) v4i* gptr_v4i;     // global (device) int4*
typedef __attribute__((address_space(3))) v4i* lptr_v4i;     // LDS int4*

constexpr int Bsz = 2, Ssz = 4096, Hsz = 16, DK = 128, DV = 128;
constexpr int CKC  = 64;             // chunk length
constexpr int NCH  = Ssz / CKC;      // 64 chunks per (b,h)
constexpr int BH   = Bsz * Hsz;      // 32
constexpr int NCID = BH * NCH;       // 2048 chunks total

constexpr int SB  = 136;  // ushort row stride for [*][128] (+8 pad; 272B rows stay 16B-aligned)
constexpr int STB = 72;   // ushort row stride for [*][64]
constexpr int SF  = 68;   // float  row stride for [64][64]
constexpr int SST = 132;  // float  row stride for state [128][128]

#if defined(__AMDGCN__) && __has_builtin(__builtin_amdgcn_global_load_async_to_lds_b128) && \
    __has_builtin(__builtin_amdgcn_s_wait_asynccnt)
#define USE_ASYNC_LDS 1
#else
#define USE_ASYNC_LDS 0
#endif

__device__ __forceinline__ unsigned short f2bf(float x) {
  union { float f; unsigned u; } a; a.f = x;
  unsigned r = a.u + 0x7FFFu + ((a.u >> 16) & 1u);   // round-to-nearest-even
  return (unsigned short)(r >> 16);
}
__device__ __forceinline__ float bf2f(unsigned short s) {
  union { unsigned u; float f; } a; a.u = ((unsigned)s) << 16; return a.f;
}

// WMMA operand fragment loader (wave32 16x16x32 bf16 layout).
// Lanes 0-15: M=lane, K {0..7,16..23}; lanes 16-31: M=lane-16, K {8..15,24..31}.
__device__ __forceinline__ bf16x16 ld_frag(const unsigned short* p, int stride, int lane) {
  const unsigned short* r = p + (lane & 15) * stride + ((lane >> 4) << 3);
  union { unsigned u[8]; bf16x16 v; } x;
  const unsigned* lo = (const unsigned*)r;
  const unsigned* hi = (const unsigned*)(r + 16);
#pragma unroll
  for (int i = 0; i < 4; ++i) { x.u[i] = lo[i]; x.u[4 + i] = hi[i]; }
  return x.v;
}

#define WMMA_BF16(A_, B_, C_) \
  __builtin_amdgcn_wmma_f32_16x16x32_bf16(false, (A_), false, (B_), (short)0, (C_), false, false)

// Stage one chunk of u/kcd (bf16, 16KB each) from workspace into LDS.
// Async path: 8 GLOBAL_LOAD_ASYNC_TO_LDS_B128 per wave, tracked by ASYNCcnt.
__device__ __forceinline__ void stage_ukcd(const unsigned short* __restrict__ Uws,
                                           const unsigned short* __restrict__ Cws,
                                           long cid, unsigned short* ub,
                                           unsigned short* cb, int tid) {
  const unsigned short* su = Uws + cid * (long)(CKC * DV);
  const unsigned short* sc = Cws + cid * (long)(CKC * DK);
#if USE_ASYNC_LDS
#pragma unroll
  for (int j2 = tid, it = 0; it < 4; ++it, j2 += 256) {
    int rr = j2 >> 4, co = (j2 & 15) * 8;
    __builtin_amdgcn_global_load_async_to_lds_b128(
        (gptr_v4i)(su + j2 * 8), (lptr_v4i)(ub + rr * SB + co), 0, 0);
    __builtin_amdgcn_global_load_async_to_lds_b128(
        (gptr_v4i)(sc + j2 * 8), (lptr_v4i)(cb + rr * SB + co), 0, 0);
  }
#else
  const uint4* pu4 = (const uint4*)su;
  const uint4* pc4 = (const uint4*)sc;
#pragma unroll
  for (int j2 = tid, it = 0; it < 4; ++it, j2 += 256) {
    int rr = j2 >> 4, co = (j2 & 15) * 8;
    uint4 xu = pu4[j2], xc = pc4[j2];
    unsigned* pu = (unsigned*)(ub + rr * SB + co);
    unsigned* pc = (unsigned*)(cb + rr * SB + co);
    pu[0] = xu.x; pu[1] = xu.y; pu[2] = xu.z; pu[3] = xu.w;
    pc[0] = xc.x; pc[1] = xc.y; pc[2] = xc.z; pc[3] = xc.w;
  }
#endif
}
__device__ __forceinline__ void wait_stage_pending8() {
#if USE_ASYNC_LDS
  __builtin_amdgcn_s_wait_asynccnt(8);   // newest 8 = next chunk; current chunk done
#endif
}
__device__ __forceinline__ void wait_stage_all() {
#if USE_ASYNC_LDS
  __builtin_amdgcn_s_wait_asynccnt(0);
#endif
}

// ---------------------------------------------------------------------------
// Phase 1: one block per chunk.  Computes u = Tinv @ (v*beta),
// kcd = Tinv @ (k*beta*exp(gcs)) where Tinv = (I + strict_tril(kb k^T * decay))^-1.
// ---------------------------------------------------------------------------
__global__ __launch_bounds__(256) void gdn_chunk_pre(
    const float* __restrict__ Kp, const float* __restrict__ Vp,
    const float* __restrict__ Gp, const float* __restrict__ Bp,
    unsigned short* __restrict__ Uws, unsigned short* __restrict__ Cws) {
  __shared__ __align__(16) unsigned short kn_s[CKC * SB];   // k normalized       [c][d]
  __shared__ __align__(16) unsigned short kb_s[CKC * SB];   // k*beta             [c][d]
  __shared__ __align__(16) unsigned short vbT_s[DV * STB];  // (v*beta)^T         [v][c]
  __shared__ __align__(16) unsigned short kgT_s[DK * STB];  // (kb*exp(gcs))^T    [d][c]
  __shared__ __align__(16) unsigned short Tb_s[CKC * STB];  // Tinv bf16          [c][d]
  __shared__ float A_s[CKC * SF];                           // strict-lower A     [i][k]
  __shared__ float Tc_s[CKC * SF];                          // Tinv columns       [col][row]
  __shared__ float gcs_s[CKC], eg_s[CKC], ieg_s[CKC];
  __shared__ float bet_s[CKC], red_s[CKC * 4], rn_s[CKC];

  const int tid = threadIdx.x;
  const int cid = blockIdx.x;
  const int bh = cid / NCH, n = cid % NCH;
  const int b = bh / Hsz, h = bh % Hsz;
  const long srow0 = (long)b * Ssz + (long)n * CKC;

  if (tid < CKC) {
    long s = srow0 + tid;
    gcs_s[tid] = Gp[s * Hsz + h];
    bet_s[tid] = Bp[s * Hsz + h];
  }
  __syncthreads();
  if (tid == 0) {                       // within-chunk cumulative log-decay
    float run = 0.f;
    for (int i = 0; i < CKC; ++i) { run += gcs_s[i]; gcs_s[i] = run; }
  }
  __syncthreads();
  if (tid < CKC) {                      // decay factors (gcs <= 0: no overflow)
    float gv = gcs_s[tid];
    eg_s[tid] = __expf(gv);
    ieg_s[tid] = __expf(-gv);
  }

  const int row = tid >> 2, sub = tid & 3;
  // ---- k rows: l2norm, beta, exp(gcs) variants
  {
    const float* kp = Kp + ((srow0 + row) * Hsz + h) * (long)DK + sub * 32;
    float v[32]; float ss = 0.f;
#pragma unroll
    for (int i = 0; i < 32; ++i) { v[i] = kp[i]; ss += v[i] * v[i]; }
    red_s[row * 4 + sub] = ss;
    __syncthreads();
    if (sub == 0)
      rn_s[row] = rsqrtf(red_s[row * 4] + red_s[row * 4 + 1] + red_s[row * 4 + 2] +
                         red_s[row * 4 + 3] + 1e-6f);
    __syncthreads();
    float rn = rn_s[row], bb = bet_s[row], egr = eg_s[row];
#pragma unroll
    for (int i = 0; i < 32; ++i) {
      float kn = v[i] * rn, kb = kn * bb;
      int d = sub * 32 + i;
      kn_s[row * SB + d] = f2bf(kn);
      kb_s[row * SB + d] = f2bf(kb);
      kgT_s[d * STB + row] = f2bf(kb * egr);
    }
  }
  // ---- v rows: beta, stored transposed
  {
    const float* vp = Vp + ((srow0 + row) * Hsz + h) * (long)DV + sub * 32;
    float bb = bet_s[row];
#pragma unroll
    for (int i = 0; i < 32; ++i) {
      int d = sub * 32 + i;
      vbT_s[d * STB + row] = f2bf(vp[i] * bb);
    }
  }
  __syncthreads();

  const int wv = tid >> 5, lane = tid & 31;
  const int hi8 = (lane >> 4) << 3, l15 = lane & 15;

  // ---- A = strict_tril((kb @ kn^T) * eg_i * ieg_j)          (16 tiles, 2/wave)
#pragma unroll
  for (int j = 0; j < 2; ++j) {
    int t = wv * 2 + j, tm = t >> 2, tn = t & 3;
    f32x8 acc = {};
#pragma unroll
    for (int ks = 0; ks < 4; ++ks) {
      bf16x16 a = ld_frag(&kb_s[(tm * 16) * SB + ks * 32], SB, lane);
      bf16x16 bb = ld_frag(&kn_s[(tn * 16) * SB + ks * 32], SB, lane);
      acc = WMMA_BF16(a, bb, acc);
    }
    int ci = tn * 16 + l15;
    float iegc = ieg_s[ci];
#pragma unroll
    for (int r = 0; r < 8; ++r) {
      int ri = tm * 16 + r + hi8;
      float val = acc[r] * eg_s[ri] * iegc;
      A_s[ri * SF + ci] = (ri > ci) ? val : 0.f;
    }
  }
  __syncthreads();

  // ---- Tinv: 64 independent unit-lower forward substitutions (column per thread)
  if (tid < CKC) {
    Tc_s[tid * SF + 0] = (tid == 0) ? 1.f : 0.f;
    for (int i = 1; i < CKC; ++i) {
      float s = (i == tid) ? 1.f : 0.f;
      for (int k2 = 0; k2 < i; ++k2) s -= A_s[i * SF + k2] * Tc_s[tid * SF + k2];
      Tc_s[tid * SF + i] = s;
    }
  }
  __syncthreads();
  for (int idx = tid; idx < CKC * CKC; idx += 256) {
    int c2 = idx >> 6, d2 = idx & 63;
    Tb_s[c2 * STB + d2] = f2bf(Tc_s[d2 * SF + c2]);
  }
  __syncthreads();

  // ---- u = Tinv @ vb ; kcd = Tinv @ (kb*exp(gcs))   (32 tiles each, 4/wave)
#pragma unroll
  for (int j = 0; j < 4; ++j) {
    int t = wv * 4 + j, tm = t >> 3, tn = t & 7;
    f32x8 au = {}, ac = {};
#pragma unroll
    for (int ks = 0; ks < 2; ++ks) {
      bf16x16 a = ld_frag(&Tb_s[(tm * 16) * STB + ks * 32], STB, lane);
      bf16x16 bv = ld_frag(&vbT_s[(tn * 16) * STB + ks * 32], STB, lane);
      bf16x16 bk = ld_frag(&kgT_s[(tn * 16) * STB + ks * 32], STB, lane);
      au = WMMA_BF16(a, bv, au);
      ac = WMMA_BF16(a, bk, ac);
    }
    long base = (long)cid * CKC * DV;
#pragma unroll
    for (int r = 0; r < 8; ++r) {
      int ri = tm * 16 + r + hi8, ci = tn * 16 + l15;
      Uws[base + ri * DV + ci] = f2bf(au[r]);
      Cws[base + ri * DK + ci] = f2bf(ac[r]);
    }
  }
}

// ---------------------------------------------------------------------------
// Phase 2: one block per (b,h); sequential scan over chunks carrying fp32
// state S^T [Dv][Dk] in LDS (bf16 shadow refreshed once per chunk).
// u/kcd staging double-buffered with async-to-LDS; q/k prefetched one chunk ahead.
// ---------------------------------------------------------------------------
__global__ __launch_bounds__(256) void gdn_scan(
    const float* __restrict__ Qp, const float* __restrict__ Kp,
    const float* __restrict__ Gp,
    const unsigned short* __restrict__ Uws, const unsigned short* __restrict__ Cws,
    float* __restrict__ Out) {
  __shared__ float ST_s[DV * SST];                             // state^T [v][k], fp32 master
  __shared__ __align__(16) unsigned short STb_s[DV * SB];      // state^T bf16 shadow
  __shared__ __align__(16) unsigned short qn_s[CKC * SB];      // q norm * Dk^-0.5   [c][d]
  __shared__ __align__(16) unsigned short kn_s[CKC * SB];      // k norm             [c][d]
  __shared__ __align__(16) unsigned short u_s[2][CKC * SB];    // u (double buffer)  [c][v]
  __shared__ __align__(16) unsigned short kc_s[2][CKC * SB];   // kcd (double buf)   [c][d]
  __shared__ __align__(16) unsigned short vnT_s[DV * STB];     // v_new^T            [v][c]
  __shared__ __align__(16) unsigned short kdT_s[DK * STB];     // k_dec^T            [d][c]
  __shared__ __align__(16) unsigned short at_s[CKC * STB];     // attn               [c][d]
  __shared__ float gcs_s[CKC], eg_s[CKC], ieg_s[CKC];
  __shared__ float red_s[CKC * 4], rn_s[CKC];
  __shared__ float glast_s;

  const int tid = threadIdx.x;
  const int bh = blockIdx.x;
  const int b = bh / Hsz, h = bh % Hsz;
  const int wv = tid >> 5, lane = tid & 31;
  const int hi8 = (lane >> 4) << 3, l15 = lane & 15;
  const int row = tid >> 2, sub = tid & 3;
  const float qscale = 0.08838834764831845f;  // 128^-0.5

  // kick off stage of chunk 0 while we clear the state
  stage_ukcd(Uws, Cws, (long)bh * NCH, u_s[0], kc_s[0], tid);

  for (int i = tid; i < DV * SST; i += 256) ST_s[i] = 0.f;
  for (int i = tid; i < DV * SB; i += 256) STb_s[i] = 0;
  __syncthreads();

  for (int n = 0; n < NCH; ++n) {
    const int cid = bh * NCH + n;
    const long srow0 = (long)b * Ssz + (long)n * CKC;
    const unsigned short* ub = u_s[n & 1];
    const unsigned short* cb = kc_s[n & 1];

    // ---- issue next chunk's async stage + prefetch next q/k rows
    if (n + 1 < NCH) {
      stage_ukcd(Uws, Cws, (long)cid + 1, u_s[(n + 1) & 1], kc_s[(n + 1) & 1], tid);
      const float* qpn = Qp + ((srow0 + CKC + row) * Hsz + h) * (long)DK + sub * 32;
      const float* kpn = Kp + ((srow0 + CKC + row) * Hsz + h) * (long)DK + sub * 32;
      __builtin_prefetch(qpn, 0, 1);
      __builtin_prefetch(kpn, 0, 1);
    }

    if (tid < CKC) gcs_s[tid] = Gp[(srow0 + tid) * Hsz + h];
    __syncthreads();
    if (tid == 0) {
      float run = 0.f;
      for (int i = 0; i < CKC; ++i) { run += gcs_s[i]; gcs_s[i] = run; }
      glast_s = run;
    }
    __syncthreads();
    if (tid < CKC) {                      // decay factors
      float gv = gcs_s[tid];
      eg_s[tid] = __expf(gv);
      ieg_s[tid] = __expf(-gv);
    }
    const float eglast = __expf(glast_s);

    // ---- q rows: l2norm * scale
    {
      const float* qp = Qp + ((srow0 + row) * Hsz + h) * (long)DK + sub * 32;
      float v[32]; float ss = 0.f;
#pragma unroll
      for (int i = 0; i < 32; ++i) { v[i] = qp[i]; ss += v[i] * v[i]; }
      red_s[row * 4 + sub] = ss;
      __syncthreads();
      if (sub == 0)
        rn_s[row] = rsqrtf(red_s[row * 4] + red_s[row * 4 + 1] + red_s[row * 4 + 2] +
                           red_s[row * 4 + 3] + 1e-6f);
      __syncthreads();
      float rn = rn_s[row] * qscale;
#pragma unroll
      for (int i = 0; i < 32; ++i) qn_s[row * SB + sub * 32 + i] = f2bf(v[i] * rn);
    }
    // ---- k rows: l2norm; k_dec^T = (kn * exp(g_last - gcs))^T
    {
      const float* kp = Kp + ((srow0 + row) * Hsz + h) * (long)DK + sub * 32;
      float v[32]; float ss = 0.f;
#pragma unroll
      for (int i = 0; i < 32; ++i) { v[i] = kp[i]; ss += v[i] * v[i]; }
      __syncthreads();                 // protect red_s reuse
      red_s[row * 4 + sub] = ss;
      __syncthreads();
      if (sub == 0)
        rn_s[row] = rsqrtf(red_s[row * 4] + red_s[row * 4 + 1] + red_s[row * 4 + 2] +
                           red_s[row * 4 + 3] + 1e-6f);
      __syncthreads();
      float rn = rn_s[row], ed = eglast * ieg_s[row];
#pragma unroll
      for (int i = 0; i < 32; ++i) {
        float kn = v[i] * rn;
        int d = sub * 32 + i;
        kn_s[row * SB + d] = f2bf(kn);
        kdT_s[d * STB + row] = f2bf(kn * ed);
      }
    }
    // current chunk's stage must have landed (newest 8 async = next chunk's)
    if (n + 1 < NCH) wait_stage_pending8(); else wait_stage_all();
    __syncthreads();

    // ---- w = kcd @ S ; v_new = u - w  (stored transposed)    32 tiles, 4/wave
#pragma unroll
    for (int j = 0; j < 4; ++j) {
      int t = wv * 4 + j, tm = t >> 3, tn = t & 7;
      f32x8 acc = {};
#pragma unroll
      for (int ks = 0; ks < 4; ++ks) {
        bf16x16 a = ld_frag(&cb[(tm * 16) * SB + ks * 32], SB, lane);
        bf16x16 bb = ld_frag(&STb_s[(tn * 16) * SB + ks * 32], SB, lane);
        acc = WMMA_BF16(a, bb, acc);
      }
#pragma unroll
      for (int r = 0; r < 8; ++r) {
        int ri = tm * 16 + r + hi8, ci = tn * 16 + l15;
        float vn = bf2f(ub[ri * SB + ci]) - acc[r];
        vnT_s[ci * STB + ri] = f2bf(vn);
      }
    }
    // ---- attn = tril((q @ k^T) * eg_i * ieg_j)               16 tiles, 2/wave
#pragma unroll
    for (int j = 0; j < 2; ++j) {
      int t = wv * 2 + j, tm = t >> 2, tn = t & 3;
      f32x8 acc = {};
#pragma unroll
      for (int ks = 0; ks < 4; ++ks) {
        bf16x16 a = ld_frag(&qn_s[(tm * 16) * SB + ks * 32], SB, lane);
        bf16x16 bb = ld_frag(&kn_s[(tn * 16) * SB + ks * 32], SB, lane);
        acc = WMMA_BF16(a, bb, acc);
      }
      int ci = tn * 16 + l15;
      float iegc = ieg_s[ci];
#pragma unroll
      for (int r = 0; r < 8; ++r) {
        int ri = tm * 16 + r + hi8;
        float val = acc[r] * eg_s[ri] * iegc;
        at_s[ri * STB + ci] = f2bf((ri >= ci) ? val : 0.f);
      }
    }
    __syncthreads();

    // ---- o = eg * (q @ S) + attn @ v_new -> global           32 tiles, 4/wave
#pragma unroll
    for (int j = 0; j < 4; ++j) {
      int t = wv * 4 + j, tm = t >> 3, tn = t & 7;
      f32x8 acc = {};
#pragma unroll
      for (int ks = 0; ks < 4; ++ks) {
        bf16x16 a = ld_frag(&qn_s[(tm * 16) * SB + ks * 32], SB, lane);
        bf16x16 bb = ld_frag(&STb_s[(tn * 16) * SB + ks * 32], SB, lane);
        acc = WMMA_BF16(a, bb, acc);
      }
#pragma unroll
      for (int r = 0; r < 8; ++r) acc[r] *= eg_s[tm * 16 + r + hi8];
#pragma unroll
      for (int ks = 0; ks < 2; ++ks) {
        bf16x16 a = ld_frag(&at_s[(tm * 16) * STB + ks * 32], STB, lane);
        bf16x16 bb = ld_frag(&vnT_s[(tn * 16) * STB + ks * 32], STB, lane);
        acc = WMMA_BF16(a, bb, acc);
      }
#pragma unroll
      for (int r = 0; r < 8; ++r) {
        int ri = tm * 16 + r + hi8, ci = tn * 16 + l15;
        Out[((srow0 + ri) * Hsz + h) * (long)DV + ci] = acc[r];
      }
    }
    // ---- state accumulators: v_new^T @ k_dec                 64 tiles, 8/wave
    f32x8 sacc[8];
#pragma unroll
    for (int j = 0; j < 8; ++j) {
      f32x8 acc = {};
#pragma unroll
      for (int ks = 0; ks < 2; ++ks) {
        bf16x16 a = ld_frag(&vnT_s[(wv * 16) * STB + ks * 32], STB, lane);
        bf16x16 bb = ld_frag(&kdT_s[(j * 16) * STB + ks * 32], STB, lane);
        acc = WMMA_BF16(a, bb, acc);
      }
      sacc[j] = acc;
    }
    __syncthreads();  // all ST_s/STb_s readers done before in-place update
#pragma unroll
    for (int j = 0; j < 8; ++j) {
#pragma unroll
      for (int r = 0; r < 8; ++r) {
        int ri = wv * 16 + r + hi8, ci = j * 16 + l15;
        float nv = ST_s[ri * SST + ci] * eglast + sacc[j][r];
        ST_s[ri * SST + ci] = nv;
        STb_s[ri * SB + ci] = f2bf(nv);
      }
    }
    __syncthreads();
  }
}

extern "C" void kernel_launch(void* const* d_in, const int* in_sizes, int n_in,
                              void* d_out, int out_size, void* d_ws, size_t ws_size,
                              hipStream_t stream) {
  (void)in_sizes; (void)n_in; (void)out_size; (void)ws_size;
  const float* q  = (const float*)d_in[0];
  const float* k  = (const float*)d_in[1];
  const float* v  = (const float*)d_in[2];
  const float* g  = (const float*)d_in[3];
  const float* bt = (const float*)d_in[4];
  float* out = (float*)d_out;
  unsigned short* uws = (unsigned short*)d_ws;                       // 32 MB
  unsigned short* cws = uws + (size_t)NCID * CKC * DV;               // 32 MB
  gdn_chunk_pre<<<NCID, 256, 0, stream>>>(k, v, g, bt, uws, cws);
  gdn_scan<<<BH, 256, 0, stream>>>(q, k, g, uws, cws, out);
}
